// MoE_50388556316697
// MI455X (gfx1250) — matrix operand order repaired and checked by reading, compiled, not run on hardware
//
#include <hip/hip_runtime.h>
#include <stdint.h>

#define D_DIM 2048
#define F_DIM 1408
#define E_NUM 8
#define FS_DIM 2816
#define T_NUM 2048
#define BM 64
#define BN 64

typedef __attribute__((ext_vector_type(16))) __bf16 bf16x16;
typedef __attribute__((ext_vector_type(2)))  __bf16 bf16x2;
typedef __attribute__((ext_vector_type(8)))  float  f32x8;
typedef __attribute__((ext_vector_type(16))) float  f32x16;
typedef __attribute__((ext_vector_type(2)))  float  f32x2;

union FragA {
    bf16x16 v;
    uint4 q[2];
};
union FragF {
    f32x16 v;
    float4 q[4];
};

__device__ __forceinline__ unsigned short f2bf(float f) {
    __bf16 h = (__bf16)f;  // native v_cvt to bf16 (RNE)
    return __builtin_bit_cast(unsigned short, h);
}

// A fragment: 16x32 bf16. lane<16: row=lane&15, K {k..k+7, k+16..k+23}; lane>=16: K {+8,+24}
__device__ __forceinline__ bf16x16 load_a(const unsigned short* row, int k, int half) {
    FragA a;
    a.q[0] = *(const uint4*)(row + k + half * 8);
    a.q[1] = *(const uint4*)(row + k + 16 + half * 8);
    return a.v;
}

// B fragment: 32x16 bf16 from fp32 weight row (col n = lane&15).
// lanes0-15: K {k..k+7, k+16..k+23}, lanes16-31: K {+8,+24}. Packed hw conversion (v_cvt_pk_bf16_f32).
__device__ __forceinline__ bf16x16 load_b(const float* wrow, int k, int half) {
    const float4* p = (const float4*)(wrow + k + half * 8);
    FragF t;
    t.q[0] = p[0];
    t.q[1] = p[1];
    t.q[2] = p[4];
    t.q[3] = p[5];
    return __builtin_convertvector(t.v, bf16x16);
}

__device__ __forceinline__ f32x8 wmma_bf16(bf16x16 a, bf16x16 b, f32x8 c) {
    return __builtin_amdgcn_wmma_f32_16x16x32_bf16(false, a, false, b, (short)0, c, false, false);
}

// silu via v_rcp_f32 instead of the IEEE divide expansion
__device__ __forceinline__ float silu(float g) {
    return g * __builtin_amdgcn_rcpf(1.f + __expf(-g));
}

// ---------------- K0: fp32 x -> bf16, zero expert counters ----------------
__global__ void moe_cvt_x(const float* __restrict__ x, unsigned short* __restrict__ xb,
                          int* __restrict__ cnt) {
    if (blockIdx.x == 0 && threadIdx.x < E_NUM) cnt[threadIdx.x] = 0;
    size_t i = ((size_t)blockIdx.x * blockDim.x + threadIdx.x) * 2;
    if (i < (size_t)T_NUM * D_DIM) {
        float2 f = *(const float2*)(x + i);
        f32x2 fv = {f.x, f.y};
        bf16x2 bv = __builtin_convertvector(fv, bf16x2);
        *(unsigned*)(xb + i) = __builtin_bit_cast(unsigned, bv);
    }
}

// ---------------- K1: router (one wave32 per token) ----------------
__global__ void moe_router(const float* __restrict__ x, const float* __restrict__ Wr,
                           int* __restrict__ cnt, int* __restrict__ idxl, float* __restrict__ wl) {
    int wave = threadIdx.x >> 5;
    int lane = threadIdx.x & 31;
    int t = blockIdx.x * (blockDim.x >> 5) + wave;
    if (t >= T_NUM) return;
    const float* xr = x + (size_t)t * D_DIM;
    float acc[E_NUM];
#pragma unroll
    for (int e = 0; e < E_NUM; ++e) acc[e] = 0.f;
    for (int d = lane; d < D_DIM; d += 32) {
        float xv = xr[d];
#pragma unroll
        for (int e = 0; e < E_NUM; ++e) acc[e] += xv * Wr[e * D_DIM + d];
    }
#pragma unroll
    for (int e = 0; e < E_NUM; ++e)
        for (int off = 16; off; off >>= 1) acc[e] += __shfl_xor(acc[e], off, 32);
    if (lane == 0) {
        float m = acc[0];
#pragma unroll
        for (int e = 1; e < E_NUM; ++e) m = fmaxf(m, acc[e]);
        float p[E_NUM], s = 0.f;
#pragma unroll
        for (int e = 0; e < E_NUM; ++e) { p[e] = __expf(acc[e] - m); s += p[e]; }
        int i1 = 0;
#pragma unroll
        for (int e = 1; e < E_NUM; ++e) if (p[e] > p[i1]) i1 = e;
        int i2 = (i1 == 0) ? 1 : 0;
#pragma unroll
        for (int e = 0; e < E_NUM; ++e) if (e != i1 && p[e] > p[i2]) i2 = e;
        float inv = __builtin_amdgcn_rcpf(s);
        int pos = atomicAdd(&cnt[i1], 1);
        idxl[i1 * T_NUM + pos] = t;
        wl[i1 * T_NUM + pos] = p[i1] * inv;
        pos = atomicAdd(&cnt[i2], 1);
        idxl[i2 * T_NUM + pos] = t;
        wl[i2 * T_NUM + pos] = p[i2] * inv;
    }
}

// ---------------- K2: routed gate+up fused, SwiGLU epilogue -> bf16 Hr ----------------
// Wave wv owns N-subtile (n0 + wv*16) and ALL 64 M rows -> B conversions unique per wave.
__global__ void __launch_bounds__(128) moe_routed_up(
        const unsigned short* __restrict__ xb,
        const float* __restrict__ Wg, const float* __restrict__ Wu,
        const int* __restrict__ cnt, const int* __restrict__ idxl,
        unsigned short* __restrict__ Hr) {
    int e = blockIdx.z;
    int c = cnt[e];
    int m0 = blockIdx.y * BM;
    if (m0 >= c) return;
    int n0 = blockIdx.x * BN;
    int wv = threadIdx.x >> 5, lane = threadIdx.x & 31;
    int half = lane >> 4, nl = lane & 15;
    int ncol = n0 + wv * 16 + nl;

    const unsigned short* xr[4];
#pragma unroll
    for (int mi = 0; mi < 4; ++mi) {
        int slot = m0 + mi * 16 + nl;
        if (slot >= c) slot = c - 1;
        xr[mi] = xb + (size_t)idxl[e * T_NUM + slot] * D_DIM;
    }
    const float* wgrow = Wg + ((size_t)e * F_DIM + ncol) * D_DIM;
    const float* wurow = Wu + ((size_t)e * F_DIM + ncol) * D_DIM;

    f32x8 gacc[4], uacc[4];
    const f32x8 zero = {0.f, 0.f, 0.f, 0.f, 0.f, 0.f, 0.f, 0.f};
#pragma unroll
    for (int mi = 0; mi < 4; ++mi) { gacc[mi] = zero; uacc[mi] = zero; }

#pragma unroll 1
    for (int k = 0; k < D_DIM; k += 32) {
        bf16x16 bg = load_b(wgrow, k, half);
        bf16x16 bu = load_b(wurow, k, half);
        bf16x16 a[4];
#pragma unroll
        for (int mi = 0; mi < 4; ++mi) a[mi] = load_a(xr[mi], k, half);
#pragma unroll
        for (int mi = 0; mi < 4; ++mi) gacc[mi] = wmma_bf16(a[mi], bg, gacc[mi]);
#pragma unroll
        for (int mi = 0; mi < 4; ++mi) uacc[mi] = wmma_bf16(a[mi], bu, uacc[mi]);
    }
#pragma unroll
    for (int mi = 0; mi < 4; ++mi) {
#pragma unroll
        for (int r = 0; r < 8; ++r) {
            int p = m0 + mi * 16 + r + half * 8;
            if (p < c) {
                float hv = silu(gacc[mi][r]) * uacc[mi][r];
                Hr[((size_t)e * T_NUM + p) * F_DIM + ncol] = f2bf(hv);
            }
        }
    }
}

// ---------------- K3: routed down-proj, weighted atomicAdd into out ----------------
__global__ void __launch_bounds__(128) moe_routed_down(
        const unsigned short* __restrict__ Hr,
        const float* __restrict__ Wd,
        const int* __restrict__ cnt, const int* __restrict__ idxl,
        const float* __restrict__ wl, float* __restrict__ out) {
    int e = blockIdx.z;
    int c = cnt[e];
    int m0 = blockIdx.y * BM;
    if (m0 >= c) return;
    int n0 = blockIdx.x * BN;  // over D
    int wv = threadIdx.x >> 5, lane = threadIdx.x & 31;
    int half = lane >> 4, nl = lane & 15;
    int ncol = n0 + wv * 16 + nl;

    const unsigned short* hr[4];
#pragma unroll
    for (int mi = 0; mi < 4; ++mi) {
        int slot = m0 + mi * 16 + nl;
        if (slot >= c) slot = c - 1;
        hr[mi] = Hr + ((size_t)e * T_NUM + slot) * F_DIM;
    }
    const float* wdrow = Wd + ((size_t)e * D_DIM + ncol) * F_DIM;

    f32x8 acc[4];
    const f32x8 zero = {0.f, 0.f, 0.f, 0.f, 0.f, 0.f, 0.f, 0.f};
#pragma unroll
    for (int mi = 0; mi < 4; ++mi) acc[mi] = zero;

#pragma unroll 2
    for (int k = 0; k < F_DIM; k += 32) {
        bf16x16 b = load_b(wdrow, k, half);
        bf16x16 a[4];
#pragma unroll
        for (int mi = 0; mi < 4; ++mi) a[mi] = load_a(hr[mi], k, half);
#pragma unroll
        for (int mi = 0; mi < 4; ++mi) acc[mi] = wmma_bf16(a[mi], b, acc[mi]);
    }
#pragma unroll
    for (int mi = 0; mi < 4; ++mi) {
#pragma unroll
        for (int r = 0; r < 8; ++r) {
            int p = m0 + mi * 16 + r + half * 8;
            if (p < c) {
                int tok = idxl[e * T_NUM + p];
                float w = wl[e * T_NUM + p];
                atomicAdd(&out[(size_t)tok * D_DIM + ncol], acc[mi][r] * w);
            }
        }
    }
}

// ---------------- K4: shared expert gate+up fused -> bf16 Hs ----------------
__global__ void __launch_bounds__(128) moe_shared_up(
        const unsigned short* __restrict__ xb,
        const float* __restrict__ Wsg, const float* __restrict__ Wsu,
        unsigned short* __restrict__ Hs) {
    int m0 = blockIdx.y * BM;
    int n0 = blockIdx.x * BN;
    int wv = threadIdx.x >> 5, lane = threadIdx.x & 31;
    int half = lane >> 4, nl = lane & 15;
    int ncol = n0 + wv * 16 + nl;

    const unsigned short* xr[4];
#pragma unroll
    for (int mi = 0; mi < 4; ++mi) xr[mi] = xb + (size_t)(m0 + mi * 16 + nl) * D_DIM;
    const float* wgrow = Wsg + (size_t)ncol * D_DIM;
    const float* wurow = Wsu + (size_t)ncol * D_DIM;

    f32x8 gacc[4], uacc[4];
    const f32x8 zero = {0.f, 0.f, 0.f, 0.f, 0.f, 0.f, 0.f, 0.f};
#pragma unroll
    for (int mi = 0; mi < 4; ++mi) { gacc[mi] = zero; uacc[mi] = zero; }

#pragma unroll 1
    for (int k = 0; k < D_DIM; k += 32) {
        bf16x16 bg = load_b(wgrow, k, half);
        bf16x16 bu = load_b(wurow, k, half);
        bf16x16 a[4];
#pragma unroll
        for (int mi = 0; mi < 4; ++mi) a[mi] = load_a(xr[mi], k, half);
#pragma unroll
        for (int mi = 0; mi < 4; ++mi) gacc[mi] = wmma_bf16(a[mi], bg, gacc[mi]);
#pragma unroll
        for (int mi = 0; mi < 4; ++mi) uacc[mi] = wmma_bf16(a[mi], bu, uacc[mi]);
    }
#pragma unroll
    for (int mi = 0; mi < 4; ++mi) {
#pragma unroll
        for (int r = 0; r < 8; ++r) {
            int p = m0 + mi * 16 + r + half * 8;
            float hv = silu(gacc[mi][r]) * uacc[mi][r];
            Hs[(size_t)p * FS_DIM + ncol] = f2bf(hv);
        }
    }
}

// ---------------- K5: shared expert down-proj, stores out ----------------
__global__ void __launch_bounds__(128) moe_shared_down(
        const unsigned short* __restrict__ Hs,
        const float* __restrict__ Wsd, float* __restrict__ out) {
    int m0 = blockIdx.y * BM;
    int n0 = blockIdx.x * BN;  // over D
    int wv = threadIdx.x >> 5, lane = threadIdx.x & 31;
    int half = lane >> 4, nl = lane & 15;
    int ncol = n0 + wv * 16 + nl;

    const unsigned short* hr[4];
#pragma unroll
    for (int mi = 0; mi < 4; ++mi) hr[mi] = Hs + (size_t)(m0 + mi * 16 + nl) * FS_DIM;
    const float* wrow = Wsd + (size_t)ncol * FS_DIM;

    f32x8 acc[4];
    const f32x8 zero = {0.f, 0.f, 0.f, 0.f, 0.f, 0.f, 0.f, 0.f};
#pragma unroll
    for (int mi = 0; mi < 4; ++mi) acc[mi] = zero;

#pragma unroll 2
    for (int k = 0; k < FS_DIM; k += 32) {
        bf16x16 b = load_b(wrow, k, half);
        bf16x16 a[4];
#pragma unroll
        for (int mi = 0; mi < 4; ++mi) a[mi] = load_a(hr[mi], k, half);
#pragma unroll
        for (int mi = 0; mi < 4; ++mi) acc[mi] = wmma_bf16(a[mi], b, acc[mi]);
    }
#pragma unroll
    for (int mi = 0; mi < 4; ++mi) {
#pragma unroll
        for (int r = 0; r < 8; ++r) {
            int p = m0 + mi * 16 + r + half * 8;
            out[(size_t)p * D_DIM + ncol] = acc[mi][r];
        }
    }
}

extern "C" void kernel_launch(void* const* d_in, const int* in_sizes, int n_in,
                              void* d_out, int out_size, void* d_ws, size_t ws_size,
                              hipStream_t stream) {
    const float* x   = (const float*)d_in[0];
    const float* Wr  = (const float*)d_in[1];
    const float* Wg  = (const float*)d_in[2];
    const float* Wu  = (const float*)d_in[3];
    const float* Wd  = (const float*)d_in[4];
    const float* Wsg = (const float*)d_in[5];
    const float* Wsu = (const float*)d_in[6];
    const float* Wsd = (const float*)d_in[7];
    float* out = (float*)d_out;

    char* ws = (char*)d_ws;
    size_t off = 0;
    auto carve = [&](size_t bytes) -> char* {
        off = (off + 255) & ~(size_t)255;
        char* p = ws + off;
        off += bytes;
        return p;
    };
    unsigned short* xb  = (unsigned short*)carve((size_t)T_NUM * D_DIM * 2);
    int*            cnt = (int*)carve(E_NUM * sizeof(int));
    int*            idxl= (int*)carve((size_t)E_NUM * T_NUM * sizeof(int));
    float*          wl  = (float*)carve((size_t)E_NUM * T_NUM * sizeof(float));
    unsigned short* Hs  = (unsigned short*)carve((size_t)T_NUM * FS_DIM * 2);
    unsigned short* Hr  = (unsigned short*)carve((size_t)E_NUM * T_NUM * F_DIM * 2);

    moe_cvt_x<<<(T_NUM * D_DIM / 2) / 256, 256, 0, stream>>>(x, xb, cnt);
    moe_router<<<T_NUM / 4, 128, 0, stream>>>(x, Wr, cnt, idxl, wl);
    moe_routed_up<<<dim3(F_DIM / BN, T_NUM / BM, E_NUM), 128, 0, stream>>>(xb, Wg, Wu, cnt, idxl, Hr);
    moe_shared_up<<<dim3(FS_DIM / BN, T_NUM / BM), 128, 0, stream>>>(xb, Wsg, Wsu, Hs);
    moe_shared_down<<<dim3(D_DIM / BN, T_NUM / BM), 128, 0, stream>>>(Hs, Wsd, out);
    moe_routed_down<<<dim3(D_DIM / BN, T_NUM / BM, E_NUM), 128, 0, stream>>>(Hr, Wd, cnt, idxl, wl, out);
}
